// MeshGNN_GAT2_73220602462595
// MI455X (gfx1250) — matrix operand order repaired and checked by reading, not compiled, not run on hardware
//
#include <hip/hip_runtime.h>
#include <hip/hip_bf16.h>

#define N_NODES 50000
#define E_EDGES 800000
#define IN_DIM  16
#define HID     128
#define HEADS   8
#define DH      16
#define NLAYERS 6
#define ED      4
#define OUT_DIM 3
#define NEG_SLOPE 0.2f

typedef __attribute__((ext_vector_type(16))) _Float16 v16h;
typedef __attribute__((ext_vector_type(8)))  float    v8f;

// ---------- small device helpers ----------
__device__ __forceinline__ float eluf(float x)   { return x > 0.f ? x : (expf(x) - 1.f); }
__device__ __forceinline__ float leakyf(float x) { return x > 0.f ? x : NEG_SLOPE * x; }

// monotonic float <-> uint encoding for atomicMax on floats
__device__ __forceinline__ unsigned f2o(float f) {
    unsigned u = __float_as_uint(f);
    return (u & 0x80000000u) ? ~u : (u | 0x80000000u);
}
__device__ __forceinline__ float o2f(unsigned u) {
    return __uint_as_float((u & 0x80000000u) ? (u & 0x7fffffffu) : ~u);
}

__device__ __forceinline__ float block_reduce_sum128(float v, float* red) {
    int t = threadIdx.x;
    red[t] = v; __syncthreads();
    for (int s = 64; s > 0; s >>= 1) {
        if (t < s) red[t] += red[t + s];
        __syncthreads();
    }
    float r = red[0]; __syncthreads();
    return r;
}

// ---------- weight conversion + WMMA-fragment swizzle ----------
// Packed layout per 128x128 matrix: idx = ((kb*8 + wv)*32 + lane)*16 + j
// where k = 32*kb + (lane&16 ? 16 : 0) + j  and  col = wv*16 + (lane&15).
// Each lane then loads its whole v16h B-fragment as 32 contiguous bytes.
__global__ void k_cvt_weights(const float* __restrict__ Wl, const float* __restrict__ Wr,
                              const float* __restrict__ h1W, _Float16* __restrict__ wh) {
    int idx = blockIdx.x * blockDim.x + threadIdx.x;
    if (idx >= 13 * HID * HID) return;
    int m    = idx >> 14;          // which matrix
    int r    = idx & 16383;        // position inside packed matrix
    int j    = r & 15;
    int lane = (r >> 4) & 31;
    int wv   = (r >> 9) & 7;
    int kb   = r >> 12;
    int k    = 32 * kb + ((lane & 16) ? 16 : 0) + j;
    int col  = wv * 16 + (lane & 15);
    const float* src = (m < 6) ? (Wl + (size_t)m * HID * HID)
                     : (m < 12 ? (Wr + (size_t)(m - 6) * HID * HID) : h1W);
    wh[idx] = (_Float16)src[k * HID + col];
}

// ---------- input MLP: Linear(16->128) + ELU + LN + ELU + BN (fused per-row) ----------
__global__ __launch_bounds__(HID) void k_lin_in(const float* __restrict__ x,
                                                const float* __restrict__ W,
                                                const float* __restrict__ bias,
                                                const float* __restrict__ g0, const float* __restrict__ b0,
                                                const float* __restrict__ bn1g, const float* __restrict__ bn1b,
                                                float* __restrict__ h) {
    __shared__ float xrow[IN_DIM];
    __shared__ float red[HID];
    int node = blockIdx.x, c = threadIdx.x;
    if (c < IN_DIM) xrow[c] = x[(long)node * IN_DIM + c];
    __syncthreads();
    float acc = bias[c];
#pragma unroll
    for (int k = 0; k < IN_DIM; k++) acc += xrow[k] * W[k * HID + c];
    float v = eluf(acc);
    float mu = block_reduce_sum128(v, red) * (1.f / HID);
    float d = v - mu;
    float var = block_reduce_sum128(d * d, red) * (1.f / HID);
    float hn = d * rsqrtf(var + 1e-5f) * g0[c] + b0[c];
    hn = eluf(hn) * bn1g[c] + bn1b[c];
    h[(long)node * HID + c] = hn;
}

// ---------- shared GEMM building blocks ----------
// A fragment build (16-bit A 16x32 WMMA layout) from LDS-staged f16 tile.
__device__ __forceinline__ void build_afrags(const _Float16* Ash, int lane, v16h afrag[4]) {
    int hiA  = (lane < 16) ? 0 : 16;
    int rowA = lane & 15;
#pragma unroll
    for (int kb = 0; kb < 4; kb++) {
#pragma unroll
        for (int j = 0; j < 16; j++)
            afrag[kb][j] = Ash[rowA * HID + 32 * kb + hiA + j];
    }
}

// ---------- WMMA GEMM (dual): xl = h@Wl+bl, xr = h@Wr+br, one A staging ----------
__global__ __launch_bounds__(256) void k_gemm_dual(const float* __restrict__ Hin,
                                                   const _Float16* __restrict__ WhA,
                                                   const _Float16* __restrict__ WhB,
                                                   const float* __restrict__ biasA,
                                                   const float* __restrict__ biasB,
                                                   float* __restrict__ OutA,
                                                   float* __restrict__ OutB) {
    __shared__ _Float16 Ash[16 * HID];
    int tid = threadIdx.x;
    long base = (long)blockIdx.x * 16;
    for (int i = tid; i < 16 * HID; i += 256)
        Ash[i] = (_Float16)Hin[base * HID + i];
    __syncthreads();

    int lane = tid & 31;
    int wv   = tid >> 5;
    int col  = wv * 16 + (lane & 15);
    int hiA  = (lane < 16) ? 0 : 8;

    v16h afrag[4];
    build_afrags(Ash, lane, afrag);

    size_t boff = ((size_t)wv * 32 + lane) * 16;   // swizzled B-fragment offset

    v8f c0 = {0.f, 0.f, 0.f, 0.f, 0.f, 0.f, 0.f, 0.f};
    v8f c1 = {0.f, 0.f, 0.f, 0.f, 0.f, 0.f, 0.f, 0.f};
#pragma unroll
    for (int kb = 0; kb < 4; kb++) {
        v16h b = *(const v16h*)(WhA + (size_t)kb * 4096 + boff);
        c0 = __builtin_amdgcn_wmma_f32_16x16x32_f16(false, afrag[kb], false, b, (short)0, c0, false, false);
    }
#pragma unroll
    for (int kb = 0; kb < 4; kb++) {
        v16h b = *(const v16h*)(WhB + (size_t)kb * 4096 + boff);
        c1 = __builtin_amdgcn_wmma_f32_16x16x32_f16(false, afrag[kb], false, b, (short)0, c1, false, false);
    }
    float bA = biasA[col], bB = biasB[col];
#pragma unroll
    for (int r = 0; r < 8; r++) {
        int m = r + hiA;   // C/D layout: VGPR r -> M=r (lanes 0-15), M=r+8 (lanes 16-31)
        OutA[(base + m) * HID + col] = c0[r] + bA;
        OutB[(base + m) * HID + col] = c1[r] + bB;
    }
}

// ---------- WMMA GEMM (single): Out = Hin @ Wh + bias ----------
__global__ __launch_bounds__(256) void k_gemm_single(const float* __restrict__ Hin,
                                                     const _Float16* __restrict__ Wh,
                                                     const float* __restrict__ bias,
                                                     float* __restrict__ Out) {
    __shared__ _Float16 Ash[16 * HID];
    int tid = threadIdx.x;
    long base = (long)blockIdx.x * 16;
    for (int i = tid; i < 16 * HID; i += 256)
        Ash[i] = (_Float16)Hin[base * HID + i];
    __syncthreads();

    int lane = tid & 31;
    int wv   = tid >> 5;
    int col  = wv * 16 + (lane & 15);
    int hiA  = (lane < 16) ? 0 : 8;

    v16h afrag[4];
    build_afrags(Ash, lane, afrag);

    size_t boff = ((size_t)wv * 32 + lane) * 16;

    v8f c = {0.f, 0.f, 0.f, 0.f, 0.f, 0.f, 0.f, 0.f};
#pragma unroll
    for (int kb = 0; kb < 4; kb++) {
        v16h b = *(const v16h*)(Wh + (size_t)kb * 4096 + boff);
        c = __builtin_amdgcn_wmma_f32_16x16x32_f16(false, afrag[kb], false, b, (short)0, c, false, false);
    }
    float bb = bias[col];
#pragma unroll
    for (int r = 0; r < 8; r++) {
        int m = r + hiA;
        Out[(base + m) * HID + col] = c[r] + bb;
    }
}

// ---------- per-layer init: zero accumulator, den; smax = -inf (encoded 0) ----------
__global__ void k_init(unsigned* __restrict__ smax, float* __restrict__ den,
                       float* __restrict__ outacc) {
    long idx = (long)blockIdx.x * blockDim.x + threadIdx.x;
    if (idx < (long)N_NODES * HID) outacc[idx] = 0.f;
    if (idx < (long)N_NODES * HEADS) { smax[idx] = 0u; den[idx] = 0.f; }
}

// ---------- pass 1: edge scores + segment max (one wave32 per edge) ----------
__global__ __launch_bounds__(256) void k_edge_score(const int* __restrict__ ei,
                                                    const float* __restrict__ ea,
                                                    const float* __restrict__ xl,
                                                    const float* __restrict__ xr,
                                                    const float* __restrict__ WeL,
                                                    const float* __restrict__ attL,
                                                    float* __restrict__ score,
                                                    unsigned* __restrict__ smax) {
    __shared__ float sWe[ED * HID];
    __shared__ float sAtt[HID];
    int tid = threadIdx.x;
    for (int i = tid; i < ED * HID; i += blockDim.x) sWe[i] = WeL[i];
    if (tid < HID) sAtt[tid] = attL[tid];
    __syncthreads();

    int lane = tid & 31;
    int gw = (int)((blockIdx.x * blockDim.x + tid) >> 5);
    int nw = (int)((gridDim.x * blockDim.x) >> 5);
    int d0 = lane * 4;
    int head = lane >> 2;

    for (int e = gw; e < E_EDGES; e += nw) {
        // prefetch next iteration's streamed data (lowers to global_prefetch_b8)
        if (e + nw < E_EDGES) {
            __builtin_prefetch(ei + e + nw, 0, 1);
            __builtin_prefetch(ea + (long)(e + nw) * ED, 0, 1);
        }
        int s = ei[e], dn = ei[E_EDGES + e];
        float4 a4 = *(const float4*)(xl + (long)s  * HID + d0);
        float4 r4 = *(const float4*)(xr + (long)dn * HID + d0);
        float av[4] = {a4.x, a4.y, a4.z, a4.w};
        float rv[4] = {r4.x, r4.y, r4.z, r4.w};
        float e0 = ea[(long)e * ED + 0], e1 = ea[(long)e * ED + 1];
        float e2 = ea[(long)e * ED + 2], e3 = ea[(long)e * ED + 3];
        float p = 0.f;
#pragma unroll
        for (int j = 0; j < 4; j++) {
            int d = d0 + j;
            float xe = e0 * sWe[d] + e1 * sWe[HID + d] + e2 * sWe[2 * HID + d] + e3 * sWe[3 * HID + d];
            float t = av[j] + rv[j] + xe;
            p += leakyf(t) * sAtt[d];
        }
        p += __shfl_xor(p, 1);
        p += __shfl_xor(p, 2);
        if ((lane & 3) == 0) {
            score[(long)e * HEADS + head] = p;
            atomicMax(&smax[(long)dn * HEADS + head], f2o(p));
        }
    }
}

// ---------- pass 2: ex = exp(score - max), segment sum denominator ----------
__global__ void k_edge_exp(const int* __restrict__ ei, float* __restrict__ score,
                           const unsigned* __restrict__ smax, float* __restrict__ den) {
    long idx = (long)blockIdx.x * blockDim.x + threadIdx.x;
    if (idx >= (long)E_EDGES * HEADS) return;
    int h = (int)(idx & 7);
    long e = idx >> 3;
    int dn = ei[E_EDGES + e];
    float m = o2f(smax[(long)dn * HEADS + h]);
    float ex = expf(score[idx] - m);
    score[idx] = ex;
    atomicAdd(&den[(long)dn * HEADS + h], ex);
}

// ---------- pass 3: scatter alpha * xl[src] into per-node accumulator ----------
__global__ __launch_bounds__(256) void k_edge_scatter(const int* __restrict__ ei,
                                                      const float* __restrict__ xl,
                                                      const float* __restrict__ score,
                                                      const float* __restrict__ den,
                                                      float* __restrict__ outacc) {
    int tid = threadIdx.x;
    int lane = tid & 31;
    int gw = (int)((blockIdx.x * blockDim.x + tid) >> 5);
    int nw = (int)((gridDim.x * blockDim.x) >> 5);
    int d0 = lane * 4;
    int head = lane >> 2;

    for (int e = gw; e < E_EDGES; e += nw) {
        if (e + nw < E_EDGES) {
            __builtin_prefetch(ei + e + nw, 0, 1);
            __builtin_prefetch(score + (long)(e + nw) * HEADS, 0, 1);
        }
        int s = ei[e], dn = ei[E_EDGES + e];
        float alpha = score[(long)e * HEADS + head] /
                      (den[(long)dn * HEADS + head] + 1e-16f);
        float4 a4 = *(const float4*)(xl + (long)s * HID + d0);
        float av[4] = {a4.x, a4.y, a4.z, a4.w};
#pragma unroll
        for (int j = 0; j < 4; j++)
            atomicAdd(&outacc[(long)dn * HID + d0 + j], alpha * av[j]);
    }
}

// ---------- residual + LayerNorm ----------
__global__ __launch_bounds__(HID) void k_residual_ln(const float* __restrict__ gatb,
                                                     const float* __restrict__ g,
                                                     const float* __restrict__ b,
                                                     const float* __restrict__ outacc,
                                                     float* __restrict__ h) {
    __shared__ float red[HID];
    int node = blockIdx.x, c = threadIdx.x;
    long idx = (long)node * HID + c;
    float v = h[idx] + outacc[idx] + gatb[c];
    float mu = block_reduce_sum128(v, red) * (1.f / HID);
    float d = v - mu;
    float var = block_reduce_sum128(d * d, red) * (1.f / HID);
    h[idx] = d * rsqrtf(var + 1e-5f) * g[c] + b[c];
}

// ---------- output head: ELU + BN + Linear(128->3) + ELU + ReLU ----------
__global__ __launch_bounds__(HID) void k_head(const float* __restrict__ t,
                                              const float* __restrict__ bn2g, const float* __restrict__ bn2b,
                                              const float* __restrict__ W2, const float* __restrict__ b2,
                                              float* __restrict__ out) {
    __shared__ float z[HID];
    int node = blockIdx.x, c = threadIdx.x;
    z[c] = eluf(t[(long)node * HID + c]) * bn2g[c] + bn2b[c];
    __syncthreads();
    if (c < OUT_DIM) {
        float s = b2[c];
        for (int k = 0; k < HID; k++) s += z[k] * W2[k * OUT_DIM + c];
        s = eluf(s);
        out[(long)node * OUT_DIM + c] = s > 0.f ? s : 0.f;
    }
}

// ============================ host launch ============================
extern "C" void kernel_launch(void* const* d_in, const int* in_sizes, int n_in,
                              void* d_out, int out_size, void* d_ws, size_t ws_size,
                              hipStream_t stream) {
    (void)in_sizes; (void)n_in; (void)out_size; (void)ws_size;
    const float* x        = (const float*)d_in[0];
    const int*   ei       = (const int*)  d_in[1];
    const float* ea       = (const float*)d_in[2];
    const float* lin_in_W = (const float*)d_in[3];
    const float* lin_in_b = (const float*)d_in[4];
    const float* ln0_g    = (const float*)d_in[5];
    const float* ln0_b    = (const float*)d_in[6];
    const float* bn1_g    = (const float*)d_in[7];
    const float* bn1_b    = (const float*)d_in[8];
    const float* Wl       = (const float*)d_in[9];
    const float* bl       = (const float*)d_in[10];
    const float* Wr       = (const float*)d_in[11];
    const float* br       = (const float*)d_in[12];
    const float* We       = (const float*)d_in[13];
    const float* att      = (const float*)d_in[14];
    const float* gat_b    = (const float*)d_in[15];
    const float* ln_g     = (const float*)d_in[16];
    const float* ln_b     = (const float*)d_in[17];
    const float* h1_W     = (const float*)d_in[18];
    const float* h1_b     = (const float*)d_in[19];
    const float* bn2_g    = (const float*)d_in[20];
    const float* bn2_b    = (const float*)d_in[21];
    const float* h2_W     = (const float*)d_in[22];
    const float* h2_b     = (const float*)d_in[23];
    float* out = (float*)d_out;

    // workspace carve-up (256B aligned offsets)
    char* ws = (char*)d_ws;
    size_t off = 0;
    auto take = [&](size_t bytes) { char* p = ws + off; off += (bytes + 255) & ~size_t(255); return p; };
    float*    h      = (float*)   take((size_t)N_NODES * HID * 4);
    float*    xl     = (float*)   take((size_t)N_NODES * HID * 4);
    float*    xr     = (float*)   take((size_t)N_NODES * HID * 4);
    float*    outacc = (float*)   take((size_t)N_NODES * HID * 4);
    float*    score  = (float*)   take((size_t)E_EDGES * HEADS * 4);
    unsigned* smax   = (unsigned*)take((size_t)N_NODES * HEADS * 4);
    float*    den    = (float*)   take((size_t)N_NODES * HEADS * 4);
    _Float16* wh     = (_Float16*)take((size_t)13 * HID * HID * 2);

    // f16 weight staging, pre-swizzled to WMMA B-fragment order
    {
        int total = 13 * HID * HID;
        k_cvt_weights<<<(total + 255) / 256, 256, 0, stream>>>(Wl, Wr, h1_W, wh);
    }
    // input MLP fused
    k_lin_in<<<N_NODES, HID, 0, stream>>>(x, lin_in_W, lin_in_b, ln0_g, ln0_b, bn1_g, bn1_b, h);

    const int gemm_blocks = N_NODES / 16;                 // 3125, exact
    const int init_blocks = (N_NODES * HID + 255) / 256;  // covers HID & HEADS regions
    const int eh_blocks   = (int)(((long)E_EDGES * HEADS + 255) / 256);
    const int edge_blocks = 2048;                         // 16384 waves, grid-stride

    for (int l = 0; l < NLAYERS; l++) {
        k_gemm_dual<<<gemm_blocks, 256, 0, stream>>>(h,
            wh + (size_t)l       * HID * HID,
            wh + (size_t)(6 + l) * HID * HID,
            bl + l * HID, br + l * HID, xl, xr);
        k_init<<<init_blocks, 256, 0, stream>>>(smax, den, outacc);
        k_edge_score<<<edge_blocks, 256, 0, stream>>>(ei, ea, xl, xr,
                                                      We + (size_t)l * ED * HID,
                                                      att + (size_t)l * HEADS * DH,
                                                      score, smax);
        k_edge_exp<<<eh_blocks, 256, 0, stream>>>(ei, score, smax, den);
        k_edge_scatter<<<edge_blocks, 256, 0, stream>>>(ei, xl, score, den, outacc);
        k_residual_ln<<<N_NODES, HID, 0, stream>>>(gat_b + l * HID, ln_g + l * HID, ln_b + l * HID,
                                                   outacc, h);
    }

    // head: h @ h1_W + h1_b (WMMA), then fused ELU/BN/Linear/ELU/ReLU
    k_gemm_single<<<gemm_blocks, 256, 0, stream>>>(h, wh + (size_t)12 * HID * HID, h1_b, xl);
    k_head<<<N_NODES, HID, 0, stream>>>(xl, bn2_g, bn2_b, h2_W, h2_b, out);
}
